// AttentionLoRA_35278861370036
// MI455X (gfx1250) — matrix operand order repaired and checked by reading, not compiled, not run on hardware
//
#include <hip/hip_runtime.h>

// ---- problem constants ----
#define SB 4096   // sequence length
#define DD 1024   // model dim
#define HH 16     // heads
#define HD 64     // head dim

#define USE_ASYNC 1   // gfx1250 GLOBAL_LOAD_ASYNC_TO_LDS_* staging

typedef _Float16 half_t;
typedef __attribute__((ext_vector_type(16))) _Float16 v16h;
typedef __attribute__((ext_vector_type(8)))  float    v8f;

union FragU { v16h v; uint4 q[2]; };

// WMMA: D = A(16x32 f16) * B(32x16 f16) + C(16x16 f32)
__device__ __forceinline__ v8f wmma32(v16h a, v16h b, v8f c) {
    return __builtin_amdgcn_wmma_f32_16x16x32_f16(false, a, false, b,
                                                  (short)0, c, false, false);
}

// A-fragment: row-major 16x32 tile in LDS, row stride ldm (halfs).
__device__ __forceinline__ v16h load_frag_a(const half_t* base, int ldm) {
    const int lane = threadIdx.x & 31;
    const int r   = lane & 15;
    const int sel = lane >> 4;
    FragU u;
    u.q[0] = *reinterpret_cast<const uint4*>(base + r * ldm + sel * 8);
    u.q[1] = *reinterpret_cast<const uint4*>(base + r * ldm + sel * 8 + 16);
    return u.v;
}

// B-fragment: B staged column-major in LDS (rows indexed by N, contiguous K).
__device__ __forceinline__ v16h load_frag_b(const half_t* base, int ldm) {
    const int lane = threadIdx.x & 31;
    const int r   = lane & 15;
    const int sel = lane >> 4;
    FragU u;
    u.q[0] = *reinterpret_cast<const uint4*>(base + r * ldm + sel * 8);
    u.q[1] = *reinterpret_cast<const uint4*>(base + r * ldm + sel * 8 + 16);
    return u.v;
}

// ---- async global -> LDS staging (16B per lane) ----
__device__ __forceinline__ void stage_b128(const half_t* g, half_t* l) {
#if USE_ASYNC
    asm volatile("global_load_async_to_lds_b128 %0, %1, off"
                 :: "v"((unsigned)(uintptr_t)l), "v"(g) : "memory");
#else
    *reinterpret_cast<uint4*>(l) = *reinterpret_cast<const uint4*>(g);
#endif
}
__device__ __forceinline__ void stage_wait() {
#if USE_ASYNC
    asm volatile("s_wait_asynccnt 0" ::: "memory");
#endif
}

// ---------------- f32 -> f16 convert ----------------
__global__ __launch_bounds__(256) void cvt_f16(const float* __restrict__ in,
                                               half_t* __restrict__ out, int n) {
    int i = blockIdx.x * 256 + threadIdx.x;
    if (i < n) out[i] = (half_t)in[i];
}

// ---------------- WMMA GEMM: C = A(4096x1024) @ W^T(1024x1024) + bias ----------------
// Block tile 128x128, 8 waves as 4(M) x 2(N), wave tile 32x64, double-buffered LDS.
// mode 0: f32 row-major [m][n]
// mode 1: f16 per-head  [h][s][d]   (Q, K)
// mode 2: f16 per-head transposed [h][d][s]  (V)
__global__ __launch_bounds__(256) void gemm_xwt(const half_t* __restrict__ A,
                                                const half_t* __restrict__ W,
                                                const float* __restrict__ bias,
                                                float* __restrict__ o32,
                                                half_t* __restrict__ o16,
                                                int mode) {
    __shared__ half_t Asm[2][128 * 40];   // 128 rows x 32 (K) halfs, stride 40
    __shared__ half_t Bsm[2][128 * 40];   // 128 W-rows (N) x 32 (K), stride 40
    const int t    = threadIdx.x;
    const int lane = t & 31;
    const int wave = t >> 5;
    const int wm = wave & 3;   // 4 waves along M (32 rows each)
    const int wn = wave >> 2;  // 2 waves along N (64 cols each)
    const int row0 = blockIdx.y * 128;
    const int col0 = blockIdx.x * 128;

    const int sr = t >> 1, sh = t & 1;   // staging: 2 threads per row, 32B each
    const half_t* gA = A + (size_t)(row0 + sr) * DD + sh * 16;
    const half_t* gB = W + (size_t)(col0 + sr) * DD + sh * 16;
    const int lofs = sr * 40 + sh * 16;

    v8f c[2][4] = {};

    // prologue: stage k0 = 0 into buffer 0
    stage_b128(gA,     &Asm[0][lofs]);
    stage_b128(gA + 8, &Asm[0][lofs + 8]);
    stage_b128(gB,     &Bsm[0][lofs]);
    stage_b128(gB + 8, &Bsm[0][lofs + 8]);
    stage_wait();
    __syncthreads();

    for (int k0 = 0; k0 < DD; k0 += 32) {
        const int cur = (k0 >> 5) & 1;
        if (k0 + 32 < DD) {   // async-stage next tile while computing this one
            const int nxt = cur ^ 1;
            stage_b128(gA + k0 + 32,     &Asm[nxt][lofs]);
            stage_b128(gA + k0 + 40,     &Asm[nxt][lofs + 8]);
            stage_b128(gB + k0 + 32,     &Bsm[nxt][lofs]);
            stage_b128(gB + k0 + 40,     &Bsm[nxt][lofs + 8]);
        }

        v16h a0 = load_frag_a(&Asm[cur][(wm * 32) * 40], 40);
        v16h a1 = load_frag_a(&Asm[cur][(wm * 32 + 16) * 40], 40);
        #pragma unroll
        for (int j = 0; j < 4; ++j) {
            v16h b = load_frag_b(&Bsm[cur][(wn * 64 + j * 16) * 40], 40);
            c[0][j] = wmma32(a0, b, c[0][j]);
            c[1][j] = wmma32(a1, b, c[1][j]);
        }

        stage_wait();       // next buffer fully landed in LDS
        __syncthreads();    // and everyone is done with the current buffer
    }

    #pragma unroll
    for (int i = 0; i < 2; ++i)
    #pragma unroll
    for (int j = 0; j < 4; ++j) {
        const int tm = row0 + wm * 32 + i * 16;
        const int tn = col0 + wn * 64 + j * 16;
        const int n  = tn + (lane & 15);
        const float bv = bias[n];
        const int mbase = tm + 8 * (lane >> 4);
        #pragma unroll
        for (int r = 0; r < 8; ++r) {
            float val = c[i][j][r] + bv;
            int m = mbase + r;
            if (mode == 0) {
                o32[(size_t)m * DD + n] = val;
            } else if (mode == 1) {
                int h = n >> 6, d = n & 63;
                o16[(size_t)h * SB * HD + (size_t)m * HD + d] = (half_t)val;
            } else {
                int h = n >> 6, d = n & 63;
                o16[(size_t)h * HD * SB + (size_t)d * SB + m] = (half_t)val;
            }
        }
    }
}

// ---------------- RoPE in-place on [h][s][64] f16 ----------------
__global__ __launch_bounds__(256) void rope_k(half_t* __restrict__ tq,
                                              const float* __restrict__ cosb,
                                              const float* __restrict__ sinb) {
    int idx = blockIdx.x * 256 + threadIdx.x;   // H*S*32 threads
    int j = idx & 31;
    int s = (idx >> 5) & (SB - 1);
    int h = idx >> 17;                          // S*32 = 2^17
    size_t base = (size_t)h * SB * HD + (size_t)s * HD + 2 * j;
    float x0 = (float)tq[base];
    float x1 = (float)tq[base + 1];
    float c  = cosb[s * 32 + j];
    float sn = sinb[s * 32 + j];
    tq[base]     = (half_t)(x0 * c - x1 * sn);
    tq[base + 1] = (half_t)(x0 * sn + x1 * c);
}

// ---------------- causal flash attention ----------------
// grid: (S/128, H); block: 256 (8 waves, 16 q-rows each -> 128 q-rows per block)
__global__ __launch_bounds__(256) void fattn(const half_t* __restrict__ qh,
                                             const half_t* __restrict__ kh,
                                             const half_t* __restrict__ vth,
                                             half_t* __restrict__ oh) {
    // PQ: Q staging (128x72) at prologue, then reused as per-wave P (8 x 16x72).
    __shared__ half_t PQ[128 * 72];
    __shared__ half_t Ksm[64 * 72];
    __shared__ half_t Vsm[64 * 72];      // V^T tile: rows = d, cols = key
    const int t = threadIdx.x, lane = t & 31, w = t >> 5;
    const int qt = blockIdx.x, h = blockIdx.y;
    const size_t hbase = (size_t)h * SB * HD;
    const float SCALE = 0.125f * 1.44269504088896f;  // 1/sqrt(64) in log2 domain

    {   // stage Q tile 128x64 (2 threads per row, 32B each)
        int r = t >> 1, hf = t & 1;
        const half_t* src = qh + hbase + (size_t)(qt * 128 + r) * HD + hf * 32;
        half_t* dst = PQ + r * 72 + hf * 32;
        stage_b128(src,     dst);
        stage_b128(src + 8, dst + 8);
    }
    stage_wait();
    __syncthreads();
    v16h qf0 = load_frag_a(PQ + (w * 16) * 72, 72);       // K-dim 0..31
    v16h qf1 = load_frag_a(PQ + (w * 16) * 72 + 32, 72);  // K-dim 32..63

    float Macc[8], Lacc[8];
    v8f of[4] = {};
    #pragma unroll
    for (int r = 0; r < 8; ++r) { Macc[r] = -__builtin_inff(); Lacc[r] = 0.f; }

    const int qrow0   = qt * 128 + w * 16 + 8 * (lane >> 4);
    const int qrowtop = qt * 128 + w * 16 + 15;            // wave-uniform
    const int ncol    = lane & 15;
    const int ktmax   = 2 * qt + 1;

    for (int kt = 0; kt <= ktmax; ++kt) {
        __syncthreads();   // all waves done reading previous K/V (and own P)
        {   // stage K (64x64 row-major) + V^T (64 d-rows x 64 keys): 16B per thread each
            int r = t >> 2, q4 = t & 3;
            stage_b128(kh  + hbase + (size_t)(kt * 64 + r) * HD + q4 * 8,
                       Ksm + r * 72 + q4 * 8);
            stage_b128(vth + hbase + (size_t)r * SB + kt * 64 + q4 * 8,
                       Vsm + r * 72 + q4 * 8);
        }
        stage_wait();
        __syncthreads();

        if (kt * 64 > qrowtop) continue;   // wave fully masked (uniform)

        // scores: 16 q-rows x 64 keys
        v8f sc[4];
        #pragma unroll
        for (int nt = 0; nt < 4; ++nt) {
            v16h kb0 = load_frag_b(Ksm + (nt * 16) * 72, 72);
            v16h kb1 = load_frag_b(Ksm + (nt * 16) * 72 + 32, 72);
            v8f s = {};
            s = wmma32(qf0, kb0, s);
            s = wmma32(qf1, kb1, s);
            const int gk = kt * 64 + nt * 16 + ncol;
            #pragma unroll
            for (int r = 0; r < 8; ++r) {
                float v = s[r] * SCALE;
                if (gk > qrow0 + r) v = -1e30f;   // causal mask
                s[r] = v;
            }
            sc[nt] = s;
        }

        // online softmax (row reductions across lanes 0..15 of each half)
        float mnew[8], alpha[8], rsum[8];
        #pragma unroll
        for (int r = 0; r < 8; ++r) {
            float mx = fmaxf(fmaxf(sc[0][r], sc[1][r]), fmaxf(sc[2][r], sc[3][r]));
            mx = fmaxf(mx, __shfl_xor(mx, 1, 32));
            mx = fmaxf(mx, __shfl_xor(mx, 2, 32));
            mx = fmaxf(mx, __shfl_xor(mx, 4, 32));
            mx = fmaxf(mx, __shfl_xor(mx, 8, 32));
            float mn = fmaxf(Macc[r], mx);
            mnew[r]  = mn;
            alpha[r] = exp2f(Macc[r] - mn);
            Macc[r]  = mn;
            rsum[r]  = 0.f;
        }
        half_t* Pw = PQ + (w * 16) * 72;   // this wave's P region (aliases its Q rows)
        #pragma unroll
        for (int nt = 0; nt < 4; ++nt)
        #pragma unroll
        for (int r = 0; r < 8; ++r) {
            float p = exp2f(sc[nt][r] - mnew[r]);
            rsum[r] += p;
            Pw[(r + 8 * (lane >> 4)) * 72 + nt * 16 + ncol] = (half_t)p;
        }
        #pragma unroll
        for (int r = 0; r < 8; ++r) {
            float s = rsum[r];
            s += __shfl_xor(s, 1, 32);
            s += __shfl_xor(s, 2, 32);
            s += __shfl_xor(s, 4, 32);
            s += __shfl_xor(s, 8, 32);
            Lacc[r] = Lacc[r] * alpha[r] + s;
            #pragma unroll
            for (int dt = 0; dt < 4; ++dt) of[dt][r] *= alpha[r];
        }

        // wave-local: make this wave's P stores visible to its own frag loads
        asm volatile("s_wait_dscnt 0" ::: "memory");

        v16h pf0 = load_frag_a(Pw, 72);
        v16h pf1 = load_frag_a(Pw + 32, 72);
        #pragma unroll
        for (int dt = 0; dt < 4; ++dt) {
            v16h vb0 = load_frag_b(Vsm + (dt * 16) * 72, 72);
            v16h vb1 = load_frag_b(Vsm + (dt * 16) * 72 + 32, 72);
            of[dt] = wmma32(pf0, vb0, of[dt]);
            of[dt] = wmma32(pf1, vb1, of[dt]);
        }
    }

    // normalize + write attention output as [s][h*64+d] f16 (GEMM-ready)
    #pragma unroll
    for (int dt = 0; dt < 4; ++dt)
    #pragma unroll
    for (int r = 0; r < 8; ++r) {
        float val = of[dt][r] / Lacc[r];
        int m = qrow0 + r;
        int d = dt * 16 + ncol;
        oh[(size_t)m * DD + h * HD + d] = (half_t)val;
    }
}

// ---------------- launch ----------------
extern "C" void kernel_launch(void* const* d_in, const int* in_sizes, int n_in,
                              void* d_out, int out_size, void* d_ws, size_t ws_size,
                              hipStream_t stream) {
    const float* x    = (const float*)d_in[0];
    const float* cosb = (const float*)d_in[2];
    const float* sinb = (const float*)d_in[3];
    const float* wq   = (const float*)d_in[5];
    const float* bq   = (const float*)d_in[6];
    const float* wk   = (const float*)d_in[7];
    const float* bk   = (const float*)d_in[8];
    const float* wv   = (const float*)d_in[9];
    const float* bv   = (const float*)d_in[10];
    const float* wo   = (const float*)d_in[11];
    const float* bo   = (const float*)d_in[12];
    float* out = (float*)d_out;

    const size_t MB = 1u << 20;
    char* ws = (char*)d_ws;
    half_t* xh  = (half_t*)(ws + 0 * MB);    // 8 MiB  (4096x1024)
    half_t* wqh = (half_t*)(ws + 8 * MB);    // 2 MiB
    half_t* wkh = (half_t*)(ws + 10 * MB);
    half_t* wvh = (half_t*)(ws + 12 * MB);
    half_t* woh = (half_t*)(ws + 14 * MB);
    half_t* qh  = (half_t*)(ws + 16 * MB);   // 8 MiB [h][s][64]
    half_t* khb = (half_t*)(ws + 24 * MB);   // 8 MiB [h][s][64]
    half_t* vth = (half_t*)(ws + 32 * MB);   // 8 MiB [h][64][s]
    half_t* oh  = (half_t*)(ws + 40 * MB);   // 8 MiB [s][1024]

    const int NX = SB * DD;   // 4M
    const int NW = DD * DD;   // 1M
    cvt_f16<<<(NX + 255) / 256, 256, 0, stream>>>(x,  xh,  NX);
    cvt_f16<<<(NW + 255) / 256, 256, 0, stream>>>(wq, wqh, NW);
    cvt_f16<<<(NW + 255) / 256, 256, 0, stream>>>(wk, wkh, NW);
    cvt_f16<<<(NW + 255) / 256, 256, 0, stream>>>(wv, wvh, NW);
    cvt_f16<<<(NW + 255) / 256, 256, 0, stream>>>(wo, woh, NW);

    dim3 ggrid(DD / 128, SB / 128);  // (8, 32)
    gemm_xwt<<<ggrid, 256, 0, stream>>>(xh, wqh, bq, nullptr, qh,  1);
    gemm_xwt<<<ggrid, 256, 0, stream>>>(xh, wkh, bk, nullptr, khb, 1);
    gemm_xwt<<<ggrid, 256, 0, stream>>>(xh, wvh, bv, nullptr, vth, 2);

    const int NR = HH * SB * 32;    // 2M rope threads
    rope_k<<<NR / 256, 256, 0, stream>>>(qh,  cosb, sinb);
    rope_k<<<NR / 256, 256, 0, stream>>>(khb, cosb, sinb);

    fattn<<<dim3(SB / 128, HH), 256, 0, stream>>>(qh, khb, vth, oh);

    gemm_xwt<<<ggrid, 256, 0, stream>>>(oh, woh, bo, out, nullptr, 0);
}
